// LocalPatchAttention_25409026523678
// MI455X (gfx1250) — compile-verified, matrix operand order checked by reading, not run on hardware
//
#include <hip/hip_runtime.h>
#include <math.h>

// ---------------------------------------------------------------------------
// LocalPatchAttention for MI455X (gfx1250, wave32, WMMA).
// B=16, CQ=512, H=W=64, CV=1024, GH=GW=16, INNER=512, P=16 tokens/cell.
// Compute-bound: ~107 GFLOP of matmuls vs ~285MB HBM traffic.  All matmuls
// run on v_wmma_f32_16x16x32_f16.  Main kernel batches 2 grid cells (M=32)
// per workgroup so each L2-resident B fragment feeds 2 WMMAs, halving L2
// weight traffic (the likely limiter) vs M=16.
// ---------------------------------------------------------------------------

typedef _Float16 v16h __attribute__((ext_vector_type(16)));
typedef _Float16 v8h  __attribute__((ext_vector_type(8)));
typedef float    v8f  __attribute__((ext_vector_type(8)));

#define MAT_ELEMS (512 * 512)
#define LDA 520                 // activation LDS stride (halves): 4-bank row skew
#define LDV 1032                // v-LN activation LDS stride (halves)

// workspace layout
#define WS_OFF_WV   ((size_t)3 * MAT_ELEMS)            // halves: v_proj_w frags
#define WS_OFF_VOUT ((size_t)2621440)                  // bytes : V rows f32

// ---------------------------------------------------------------------------
// Weight swizzles: W[out=512][in=K] f32 -> f16 B-fragment layout.
// Fragment (ntile n, kstep t): lane l holds 16 halves
//   B[k][col], col = n*16+(l&15), k = t*32 + (j&7) + 16*(j>>3) + 8*(l>>4)
// stored contiguously per lane: dst[((n*KSTEPS + t)*32 + l)*16 + j].
// ---------------------------------------------------------------------------
__global__ __launch_bounds__(256)
void lpa_swizzle_w512(const float* __restrict__ w0,   // q_proj_w
                      const float* __restrict__ w1,   // channel_meanings
                      const float* __restrict__ w2,   // out_w
                      _Float16* __restrict__ dst)
{
    int id   = blockIdx.x * 256 + threadIdx.x;   // 0 .. 3*512*512-1
    int j    = id & 15;
    int lane = (id >> 4) & 31;
    int t    = (id >> 9) & 15;                   // 16 ksteps (K=512)
    int n    = (id >> 13) & 31;                  // 32 ntiles
    int mat  = id >> 18;
    const float* W = (mat == 0) ? w0 : ((mat == 1) ? w1 : w2);
    int row = n * 16 + (lane & 15);
    int col = t * 32 + (j & 7) + 16 * (j >> 3) + 8 * (lane >> 4);
    dst[id] = (_Float16)W[row * 512 + col];
}

__global__ __launch_bounds__(256)
void lpa_swizzle_wv(const float* __restrict__ W,      // v_proj_w [512x1024]
                    _Float16* __restrict__ dst)
{
    int id   = blockIdx.x * 256 + threadIdx.x;   // 0 .. 512*1024-1
    int j    = id & 15;
    int lane = (id >> 4) & 31;
    int t    = (id >> 9) & 31;                   // 32 ksteps (K=1024)
    int n    = id >> 14;                         // 32 ntiles
    int row  = n * 16 + (lane & 15);
    int col  = t * 32 + (j & 7) + 16 * (j >> 3) + 8 * (lane >> 4);
    dst[id] = (_Float16)W[row * 1024 + col];
}

// A fragment from LDS: two aligned 16B groups (aRow already includes hl*8).
__device__ __forceinline__ v16h ld_a(const _Float16* aRow, int t)
{
    v8h lo = *(const v8h*)(aRow + t * 32);
    v8h hi = *(const v8h*)(aRow + t * 32 + 16);
    return __builtin_shufflevector(lo, hi, 0,1,2,3,4,5,6,7,8,9,10,11,12,13,14,15);
}

#define WMMA16(A, Bf, C) \
    __builtin_amdgcn_wmma_f32_16x16x32_f16(false, (A), false, (Bf), (short)0, (C), false, false)

// ---------------------------------------------------------------------------
// Single-M-tile pipelined block (used by the v-projection GEMM).
// ---------------------------------------------------------------------------
template <int KSTEPS>
__device__ __forceinline__ void wmma_block(const _Float16* __restrict__ aRow,
                                           const _Float16* __restrict__ bBase,
                                           v8f acc[4])
{
    v16h a_c  = ld_a(aRow, 0);
    v16h b_c0 = *(const v16h*)(bBase + 0 * KSTEPS * 512);
    v16h b_c1 = *(const v16h*)(bBase + 1 * KSTEPS * 512);
    v16h b_c2 = *(const v16h*)(bBase + 2 * KSTEPS * 512);
    v16h b_c3 = *(const v16h*)(bBase + 3 * KSTEPS * 512);
    #pragma unroll
    for (int t = 0; t < KSTEPS; ++t) {
        v16h a_n, b_n0, b_n1, b_n2, b_n3;
        const bool has_next = (t + 1 < KSTEPS);
        if (has_next) {
            b_n0 = *(const v16h*)(bBase + 0 * KSTEPS * 512 + (t + 1) * 512);
            b_n1 = *(const v16h*)(bBase + 1 * KSTEPS * 512 + (t + 1) * 512);
            b_n2 = *(const v16h*)(bBase + 2 * KSTEPS * 512 + (t + 1) * 512);
            b_n3 = *(const v16h*)(bBase + 3 * KSTEPS * 512 + (t + 1) * 512);
            a_n  = ld_a(aRow, t + 1);
        }
        acc[0] = WMMA16(a_c, b_c0, acc[0]);
        acc[1] = WMMA16(a_c, b_c1, acc[1]);
        acc[2] = WMMA16(a_c, b_c2, acc[2]);
        acc[3] = WMMA16(a_c, b_c3, acc[3]);
        if (has_next) { a_c = a_n; b_c0 = b_n0; b_c1 = b_n1; b_c2 = b_n2; b_c3 = b_n3; }
    }
}

// ---------------------------------------------------------------------------
// Two-M-tile pipelined block (main kernel): 4 ntiles x 2 mtiles per wave;
// every B fragment feeds 2 WMMAs.  B depth-1 double-buffered; A from LDS.
// ---------------------------------------------------------------------------
template <int KSTEPS>
__device__ __forceinline__ void wmma_block2(const _Float16* __restrict__ aRow0,
                                            const _Float16* __restrict__ aRow1,
                                            const _Float16* __restrict__ bBase,
                                            v8f acc0[4], v8f acc1[4])
{
    v16h b_c0 = *(const v16h*)(bBase + 0 * KSTEPS * 512);
    v16h b_c1 = *(const v16h*)(bBase + 1 * KSTEPS * 512);
    v16h b_c2 = *(const v16h*)(bBase + 2 * KSTEPS * 512);
    v16h b_c3 = *(const v16h*)(bBase + 3 * KSTEPS * 512);
    #pragma unroll
    for (int t = 0; t < KSTEPS; ++t) {
        v16h b_n0, b_n1, b_n2, b_n3;
        const bool has_next = (t + 1 < KSTEPS);
        if (has_next) {                          // issue next B loads first
            b_n0 = *(const v16h*)(bBase + 0 * KSTEPS * 512 + (t + 1) * 512);
            b_n1 = *(const v16h*)(bBase + 1 * KSTEPS * 512 + (t + 1) * 512);
            b_n2 = *(const v16h*)(bBase + 2 * KSTEPS * 512 + (t + 1) * 512);
            b_n3 = *(const v16h*)(bBase + 3 * KSTEPS * 512 + (t + 1) * 512);
        }
        const v16h a0 = ld_a(aRow0, t);
        const v16h a1 = ld_a(aRow1, t);
        acc0[0] = WMMA16(a0, b_c0, acc0[0]);
        acc1[0] = WMMA16(a1, b_c0, acc1[0]);
        acc0[1] = WMMA16(a0, b_c1, acc0[1]);
        acc1[1] = WMMA16(a1, b_c1, acc1[1]);
        acc0[2] = WMMA16(a0, b_c2, acc0[2]);
        acc1[2] = WMMA16(a1, b_c2, acc1[2]);
        acc0[3] = WMMA16(a0, b_c3, acc0[3]);
        acc1[3] = WMMA16(a1, b_c3, acc1[3]);
        if (has_next) { b_c0 = b_n0; b_c1 = b_n1; b_c2 = b_n2; b_c3 = b_n3; }
    }
}

// ---------------------------------------------------------------------------
// Batched v-projection: LN all 4096 v tokens, then [4096x1024]@[1024x512]^T
// -> Vout[4096][512] f32.  One workgroup per 16 tokens; 256 workgroups.
// ---------------------------------------------------------------------------
__global__ __launch_bounds__(256)
void lpa_vproj(const float* __restrict__ v,      // [16,1024,16,16]
               const float* __restrict__ vng, const float* __restrict__ vnb,
               const float* __restrict__ vpb,
               const _Float16* __restrict__ wvfrag,
               float* __restrict__ Vout)         // [4096,512]
{
    extern __shared__ char smem[];
    _Float16* bufV = (_Float16*)smem;            // 16 x 1032 halves

    const int tid = threadIdx.x, wave = tid >> 5, lane = tid & 31;
    const int cell0 = blockIdx.x * 16;
    const int m_lane = lane & 15, hl = lane >> 4;

    #pragma unroll
    for (int rr = 0; rr < 2; ++rr) {
        const int r = wave * 2 + rr;
        const int cell = cell0 + r;              // (b*16+hh)*16+ww
        const int b = cell >> 8;
        const float* src = v + (size_t)b * 1024 * 256 + (cell & 255);
        float xr[32];
        float s = 0.f, ss = 0.f;
        #pragma unroll
        for (int i = 0; i < 32; ++i) {
            const int c = lane + i * 32;
            float x = src[(size_t)c * 256];
            xr[i] = x; s += x; ss += x * x;
        }
        #pragma unroll
        for (int off = 16; off; off >>= 1) { s += __shfl_xor(s, off); ss += __shfl_xor(ss, off); }
        const float mu = s * (1.f / 1024.f);
        const float rs = rsqrtf(ss * (1.f / 1024.f) - mu * mu + 1e-5f);
        #pragma unroll
        for (int i = 0; i < 32; ++i) {
            const int c = lane + i * 32;
            bufV[r * LDV + c] = (_Float16)((xr[i] - mu) * rs * vng[c] + vnb[c]);
        }
    }
    __syncthreads();

    v8f acc[4] = {};
    wmma_block<32>(bufV + m_lane * LDV + hl * 8,
                   wvfrag + ((size_t)(wave * 4 * 32) * 32 + lane) * 16, acc);

    #pragma unroll
    for (int nn = 0; nn < 4; ++nn) {
        const int i = (wave * 4 + nn) * 16 + m_lane;
        const float bias = vpb[i];
        #pragma unroll
        for (int r8 = 0; r8 < 8; ++r8) {
            const int m = r8 + 8 * hl;
            Vout[(size_t)(cell0 + m) * 512 + i] = acc[nn][r8] + bias;
        }
    }
}

// ---------------------------------------------------------------------------
// Main fused kernel: TWO grid cells (32 tokens) per workgroup; 2048 WGs.
// LN(q) -> WMMA(q_proj) -> WMMA(channel_meanings)+sigmoid*V -> WMMA(out_w)+res
// ---------------------------------------------------------------------------
__global__ __launch_bounds__(256)
void lpa_main(const float* __restrict__ q,       // [16,512,64,64]
              const float* __restrict__ qng, const float* __restrict__ qnb,
              const float* __restrict__ qpb, const float* __restrict__ outb,
              const _Float16* __restrict__ wfrag, // 3 swizzled 512x512 mats
              const float* __restrict__ Vall,     // [4096,512]
              float* __restrict__ out)            // [16,512,64,64]
{
    extern __shared__ char smem[];
    _Float16* bufA = (_Float16*)smem;            // 32*520 halves = 33280 B
    _Float16* bufB = (_Float16*)(smem + 33280);  // 32*520 halves

    const int tid = threadIdx.x, wave = tid >> 5, lane = tid & 31;
    const int bid = blockIdx.x;                  // handles cells 2*bid, 2*bid+1
    const int m_lane = lane & 15, hl = lane >> 4;

    // ---- Phase 0: layernorm of 32 query tokens (4 rows per wave) ----
    #pragma unroll
    for (int rr = 0; rr < 4; ++rr) {
        const int p    = wave * 4 + rr;          // 0..31
        const int cell = bid * 2 + (p >> 4);
        const int b = cell >> 8, hh = (cell >> 4) & 15, ww = cell & 15;
        const int m = p & 15;
        const int py = m >> 2, px = m & 3;
        const float* src = q + (((size_t)b * 512) * 64 + (hh * 4 + py)) * 64 + (ww * 4 + px);
        float xr[16];
        float s = 0.f, ss = 0.f;
        #pragma unroll
        for (int i = 0; i < 16; ++i) {
            const int c = lane + i * 32;
            float x = src[(size_t)c * 4096];
            xr[i] = x; s += x; ss += x * x;
        }
        #pragma unroll
        for (int off = 16; off; off >>= 1) { s += __shfl_xor(s, off); ss += __shfl_xor(ss, off); }
        const float mu = s * (1.f / 512.f);
        const float rs = rsqrtf(ss * (1.f / 512.f) - mu * mu + 1e-5f);
        #pragma unroll
        for (int i = 0; i < 16; ++i) {
            const int c = lane + i * 32;
            bufA[p * LDA + c] = (_Float16)((xr[i] - mu) * rs * qng[c] + qnb[c]);
        }
    }
    __syncthreads();

    const _Float16* aA0 = bufA + m_lane * LDA + hl * 8;          // mtile 0
    const _Float16* aA1 = bufA + (16 + m_lane) * LDA + hl * 8;   // mtile 1
    const _Float16* aB0 = bufB + m_lane * LDA + hl * 8;
    const _Float16* aB1 = bufB + (16 + m_lane) * LDA + hl * 8;
    const size_t blane = ((size_t)(wave * 4 * 16) * 32 + lane) * 16;

    // ---- Matmul 1: Q = qn @ q_proj_w^T + b -> bufB (f16) ----
    {
        v8f acc0[4] = {}, acc1[4] = {};
        wmma_block2<16>(aA0, aA1, wfrag + blane, acc0, acc1);
        #pragma unroll
        for (int nn = 0; nn < 4; ++nn) {
            const int i = (wave * 4 + nn) * 16 + m_lane;
            const float bias = qpb[i];
            #pragma unroll
            for (int r = 0; r < 8; ++r) {
                bufB[(r + 8 * hl) * LDA + i]      = (_Float16)(acc0[nn][r] + bias);
                bufB[(16 + r + 8 * hl) * LDA + i] = (_Float16)(acc1[nn][r] + bias);
            }
        }
    }
    __syncthreads();

    // ---- Matmul 2: logits = Q @ cm^T; x = sigmoid(logits*scale)*V -> bufA ----
    {
        const float SCALE = 0.04419417382415922f;  // 512^-0.5
        v8f acc0[4] = {}, acc1[4] = {};
        wmma_block2<16>(aB0, aB1, wfrag + MAT_ELEMS + blane, acc0, acc1);
        const float* Vrow0 = Vall + (size_t)(bid * 2) * 512;
        const float* Vrow1 = Vrow0 + 512;
        #pragma unroll
        for (int nn = 0; nn < 4; ++nn) {
            const int i = (wave * 4 + nn) * 16 + m_lane;
            const float vi0 = Vrow0[i];
            const float vi1 = Vrow1[i];
            #pragma unroll
            for (int r = 0; r < 8; ++r) {
                const float z0 = acc0[nn][r] * SCALE;
                const float z1 = acc1[nn][r] * SCALE;
                bufA[(r + 8 * hl) * LDA + i]      = (_Float16)(vi0 / (1.f + __expf(-z0)));
                bufA[(16 + r + 8 * hl) * LDA + i] = (_Float16)(vi1 / (1.f + __expf(-z1)));
            }
        }
    }
    __syncthreads();

    // ---- Matmul 3: out = x @ out_w^T + out_b + q (residual, scatter) ----
    {
        v8f acc0[4] = {}, acc1[4] = {};
        wmma_block2<16>(aA0, aA1, wfrag + 2 * MAT_ELEMS + blane, acc0, acc1);
        #pragma unroll
        for (int mt = 0; mt < 2; ++mt) {
            const int cell = bid * 2 + mt;
            const int b = cell >> 8, hh = (cell >> 4) & 15, ww = cell & 15;
            #pragma unroll
            for (int nn = 0; nn < 4; ++nn) {
                const int c = (wave * 4 + nn) * 16 + m_lane;
                const float cb = outb[c];
                #pragma unroll
                for (int r = 0; r < 8; ++r) {
                    const int m = r + 8 * hl;
                    const int py = m >> 2, px = m & 3;
                    const size_t idx = (((size_t)b * 512 + c) * 64 + (hh * 4 + py)) * 64
                                     + (ww * 4 + px);
                    const float a = mt ? acc1[nn][r] : acc0[nn][r];
                    out[idx] = a + cb + q[idx];    // q is L2-hot (134MB < 192MB L2)
                }
            }
        }
    }
}

// ---------------------------------------------------------------------------
extern "C" void kernel_launch(void* const* d_in, const int* in_sizes, int n_in,
                              void* d_out, int out_size, void* d_ws, size_t ws_size,
                              hipStream_t stream) {
    const float* q   = (const float*)d_in[0];
    const float* v   = (const float*)d_in[1];
    const float* qng = (const float*)d_in[2];
    const float* qnb = (const float*)d_in[3];
    const float* vng = (const float*)d_in[4];
    const float* vnb = (const float*)d_in[5];
    const float* qpw = (const float*)d_in[6];
    const float* qpb = (const float*)d_in[7];
    const float* vpw = (const float*)d_in[8];
    const float* vpb = (const float*)d_in[9];
    const float* cm  = (const float*)d_in[10];
    const float* ow  = (const float*)d_in[11];
    const float* ob  = (const float*)d_in[12];
    float* out = (float*)d_out;

    _Float16* wfrag  = (_Float16*)d_ws;                       // 1.5 MB
    _Float16* wvfrag = (_Float16*)d_ws + WS_OFF_WV;           // 1.0 MB
    float*    Vout   = (float*)((char*)d_ws + WS_OFF_VOUT);   // 8.0 MB

    // 1) weight swizzles (f32 -> f16 WMMA fragment layout)
    lpa_swizzle_w512<<<(3 * MAT_ELEMS) / 256, 256, 0, stream>>>(qpw, cm, ow, wfrag);
    lpa_swizzle_wv <<<(512 * 1024) / 256, 256, 0, stream>>>(vpw, wvfrag);
    // 2) batched v layernorm + projection GEMM
    lpa_vproj<<<256, 256, 16 * LDV * 2, stream>>>(v, vng, vnb, vpb, wvfrag, Vout);
    // 3) main fused kernel: two grid cells per workgroup
    lpa_main<<<2048, 256, 2 * 32 * LDA * 2, stream>>>(
        q, qng, qnb, qpb, ob, wfrag, Vout, out);
}
